// LSADecoder_24704651887140
// MI455X (gfx1250) — matrix operand order repaired and checked by reading, not compiled, not run on hardware
//
#include <hip/hip_runtime.h>
#include <hip/hip_bf16.h>
#include <stddef.h>
#include <stdint.h>

// ---------------------------------------------------------------------------
// LSTM+attention decoder for MI455X (gfx1250).
// - Per-step GEMMs: v_wmma_f32_16x16x32_bf16 (bf16 operands, f32 accum).
// - A-tiles staged to LDS via global_load_async_to_lds_b128 (ASYNCcnt path),
//   deduplicating the 4x-redundant A loads across the 4 waves of a block.
// - Weights bf16-converted once per call; ~12MB, resident in the 192MB L2.
// - Stream-ordered kernels provide the recurrence synchronization.
// ---------------------------------------------------------------------------

#define B_ 64
#define T_ 64
#define S_ 256
#define H_ 512
#define L_ 2
#define G_ 2048   // 4*H
#define O_ 3

typedef __attribute__((ext_vector_type(16))) __bf16 v16bf;
typedef __attribute__((ext_vector_type(8)))  float  v8f;

union Frag { uint4 q[2]; v16bf v; };

__device__ __forceinline__ unsigned short f2bf(float f) {
  unsigned int u = __float_as_uint(f);
  u += 0x7FFFu + ((u >> 16) & 1u);   // round-to-nearest-even
  return (unsigned short)(u >> 16);
}

__device__ __forceinline__ float sigm(float x) { return 1.0f / (1.0f + __expf(-x)); }

// A-fragment (16x32 bf16, M x K): lane holds row m=lane&15,
// elems 0..7  = K[k0 + 8*(lane>>4) .. +7], elems 8..15 = K[k0+16+8*(lane>>4) .. +7]
__device__ __forceinline__ v16bf load_a(const unsigned short* row, int k0, int lh) {
  Frag f;
  f.q[0] = *reinterpret_cast<const uint4*>(row + k0 + lh * 8);
  f.q[1] = *reinterpret_cast<const uint4*>(row + k0 + 16 + lh * 8);
  return f.v;
}

// B-fragment (32x16 bf16, K x N) built from row-major weight W[n][k]:
// lane holds col n=lane&15, elems 0..7 = K[k0 + 8*(lane>>4) .. +7], elems 8..15 = K[k0+16+8*(lane>>4) .. +7]
__device__ __forceinline__ v16bf load_b(const unsigned short* __restrict__ wrow, int k0, int lh) {
  Frag f;
  f.q[0] = *reinterpret_cast<const uint4*>(wrow + k0 + lh * 8);
  f.q[1] = *reinterpret_cast<const uint4*>(wrow + k0 + 16 + lh * 8);
  return f.v;
}

// Async copy of one 16-byte chunk from global to LDS (per-lane, ASYNCcnt).
__device__ __forceinline__ void async_cp16(unsigned lds_off, const void* gptr) {
  asm volatile("global_load_async_to_lds_b128 %0, %1, off"
               :: "v"(lds_off), "v"(gptr) : "memory");
}
__device__ __forceinline__ void async_wait() {
  asm volatile("s_wait_asynccnt 0" ::: "memory");
}
__device__ __forceinline__ unsigned lds_off_of(const void* p) {
  return (unsigned)(uintptr_t)p;   // generic->LDS: low 32 bits (aperture rule)
}

// --------------------------- conversion ------------------------------------
__global__ void cvt_bf16_kernel(const float* __restrict__ in, unsigned short* __restrict__ out, int n) {
  int i = blockIdx.x * blockDim.x + threadIdx.x;
  if (i < n) out[i] = f2bf(in[i]);
}

// --------------------------- LSTM gates GEMM -------------------------------
// gates(64 x 2048) = A1(64 x 512) @ Wih^T + Ah(64 x 512) @ Whh^T + bih + bhh
// All 4 waves of a block share one M-tile: stage both A tiles to LDS (async).
__global__ __launch_bounds__(128) void gates_kernel(
    const unsigned short* __restrict__ A1, int lda1,
    const unsigned short* __restrict__ Ah,
    const unsigned short* __restrict__ Wih,
    const unsigned short* __restrict__ Whh,
    const float* __restrict__ bih, const float* __restrict__ bhh,
    float* __restrict__ gates) {
  __shared__ uint4 shA1q[16 * 512 / 8];   // 16 rows x 512 bf16 = 16KB
  __shared__ uint4 shAhq[16 * 512 / 8];
  unsigned short* shA1 = (unsigned short*)shA1q;
  unsigned short* shAh = (unsigned short*)shAhq;

  int wave = threadIdx.x >> 5, lane = threadIdx.x & 31;
  int tile = blockIdx.x * 4 + wave;        // 512 tiles: 4 (M) x 128 (N)
  int mt = tile >> 7, nt = tile & 127;     // mt uniform within the block
  int ll = lane & 15, lh = lane >> 4;
  int tid = threadIdx.x;

  // Stage A1/Ah tiles (16 rows x 512 bf16 each): 1024 16B chunks per matrix.
#pragma unroll
  for (int c = tid; c < 1024; c += 128) {
    int row = c >> 6;                 // 64 chunks of 8 elems per row
    int col = (c & 63) * 8;
    async_cp16(lds_off_of(&shA1[row * 512 + col]),
               A1 + (size_t)(mt * 16 + row) * lda1 + col);
    async_cp16(lds_off_of(&shAh[row * 512 + col]),
               Ah + (size_t)(mt * 16 + row) * H_ + col);
  }
  const unsigned short* w1row = Wih + (size_t)(nt * 16 + ll) * H_;
  const unsigned short* w2row = Whh + (size_t)(nt * 16 + ll) * H_;
  __builtin_prefetch((const void*)w1row, 0, 2);   // global_prefetch_b8
  __builtin_prefetch((const void*)w2row, 0, 2);
  async_wait();
  __syncthreads();

  const unsigned short* a1row = shA1 + ll * 512;
  const unsigned short* ahrow = shAh + ll * 512;
  v8f acc = {};
#pragma unroll
  for (int k0 = 0; k0 < H_; k0 += 32) {
    v16bf a = load_a(a1row, k0, lh);
    v16bf b = load_b(w1row, k0, lh);
    acc = __builtin_amdgcn_wmma_f32_16x16x32_bf16(false, a, false, b, (short)0, acc, false, false);
  }
#pragma unroll
  for (int k0 = 0; k0 < H_; k0 += 32) {
    v16bf a = load_a(ahrow, k0, lh);
    v16bf b = load_b(w2row, k0, lh);
    acc = __builtin_amdgcn_wmma_f32_16x16x32_bf16(false, a, false, b, (short)0, acc, false, false);
  }
  int n = nt * 16 + ll;
  float bias = bih[n] + bhh[n];
  int mbase = mt * 16 + lh * 8;
#pragma unroll
  for (int r = 0; r < 8; ++r)
    gates[(size_t)(mbase + r) * G_ + n] = acc[r] + bias;
}

// ---------------- fused LSTM cell + inter attention (Luong) ----------------
// One block per batch element: cell elementwise -> h_raw lives in LDS for the
// attention phase; bf16 copies written for next-step gates / concat.
__global__ __launch_bounds__(256) void cell_attn_kernel(
    const float* __restrict__ gates, float* __restrict__ c,
    unsigned short* __restrict__ hraw_bf, unsigned short* __restrict__ cat1,
    const float* __restrict__ enc) {
  __shared__ float sh_h[H_];
  __shared__ float sh_w[S_];
  __shared__ float sh_r[S_];
  int b = blockIdx.x, tid = threadIdx.x;

  // LSTM cell (i,f,g,o at n, n+512, n+1024, n+1536)
  const float* g = gates + (size_t)b * G_;
#pragma unroll
  for (int half = 0; half < 2; ++half) {
    int d = tid + half * 256;
    float ig = sigm(g[d]);
    float fg = sigm(g[512 + d]);
    float gg = tanhf(g[1024 + d]);
    float og = sigm(g[1536 + d]);
    int idx = b * H_ + d;
    float cn = fg * c[idx] + ig * gg;
    c[idx] = cn;
    float h = og * tanhf(cn);
    sh_h[d] = h;
    unsigned short hb = f2bf(h);
    hraw_bf[idx] = hb;
    cat1[(size_t)b * 1024 + 512 + d] = hb;   // [ctx, h_raw] right half
  }
  __syncthreads();

  // scores over S=256 encoder positions
  const float* row = enc + ((size_t)b * S_ + tid) * H_;
  float sc = 0.f;
  for (int k = 0; k < H_; ++k) sc += sh_h[k] * row[k];
  sh_r[tid] = sc;
  __syncthreads();
  for (int off = 128; off > 0; off >>= 1) {
    if (tid < off) sh_r[tid] = fmaxf(sh_r[tid], sh_r[tid + off]);
    __syncthreads();
  }
  float mx = sh_r[0];
  __syncthreads();
  float e = __expf(sc - mx);
  sh_w[tid] = e;
  sh_r[tid] = e;
  __syncthreads();
  for (int off = 128; off > 0; off >>= 1) {
    if (tid < off) sh_r[tid] += sh_r[tid + off];
    __syncthreads();
  }
  float inv = 1.0f / sh_r[0];
#pragma unroll
  for (int half = 0; half < 2; ++half) {
    int d = tid + half * 256;
    float a = 0.f;
    for (int s = 0; s < S_; ++s) a += sh_w[s] * enc[((size_t)b * S_ + s) * H_ + d];
    cat1[(size_t)b * 1024 + d] = f2bf(a * inv);   // ctx -> left half
  }
}

// --------------------------- h_vec = tanh(cat1 @ Wint^T + b) ---------------
__global__ __launch_bounds__(128) void hvec_kernel(
    const unsigned short* __restrict__ cat1, const unsigned short* __restrict__ W,
    const float* __restrict__ bias, float* __restrict__ hvec_f,
    unsigned short* __restrict__ cat2) {
  __shared__ uint4 shAq[16 * 1024 / 8];   // 16 rows x 1024 bf16 = 32KB
  unsigned short* shA = (unsigned short*)shAq;
  int wave = threadIdx.x >> 5, lane = threadIdx.x & 31;
  int tile = blockIdx.x * 4 + wave;        // 128 tiles: 4 (M) x 32 (N)
  int mt = tile >> 5, nt = tile & 31;      // mt uniform within the block
  int ll = lane & 15, lh = lane >> 4;
  int tid = threadIdx.x;
#pragma unroll
  for (int cix = tid; cix < 2048; cix += 128) {
    int row = cix >> 7;                // 128 chunks per row
    int col = (cix & 127) * 8;
    async_cp16(lds_off_of(&shA[row * 1024 + col]),
               cat1 + (size_t)(mt * 16 + row) * 1024 + col);
  }
  const unsigned short* wrow = W + (size_t)(nt * 16 + ll) * 1024;
  __builtin_prefetch((const void*)wrow, 0, 2);
  async_wait();
  __syncthreads();

  const unsigned short* arow = shA + ll * 1024;
  v8f acc = {};
#pragma unroll
  for (int k0 = 0; k0 < 1024; k0 += 32) {
    v16bf a = load_a(arow, k0, lh);
    v16bf b = load_b(wrow, k0, lh);
    acc = __builtin_amdgcn_wmma_f32_16x16x32_bf16(false, a, false, b, (short)0, acc, false, false);
  }
  int n = nt * 16 + ll;
  float bb = bias[n];
  int mbase = mt * 16 + lh * 8;
#pragma unroll
  for (int r = 0; r < 8; ++r) {
    int m = mbase + r;
    float t = tanhf(acc[r] + bb);
    hvec_f[(size_t)m * H_ + n] = t;
    cat2[(size_t)m * 1024 + 512 + n] = f2bf(t);   // right half of [ctx2, h_vec]
  }
}

// --------------------------- causal self attention -------------------------
__global__ __launch_bounds__(256) void self_attn_kernel(
    const float* __restrict__ hvec, const float* __restrict__ pre,
    const float* __restrict__ ref, unsigned short* __restrict__ cat2, int step) {
  __shared__ float sh_h[H_];
  __shared__ float sh_w[T_];
  __shared__ float sh_r[T_];
  int b = blockIdx.x, tid = threadIdx.x;
  sh_h[tid] = hvec[b * H_ + tid];
  sh_h[tid + 256] = hvec[b * H_ + 256 + tid];
  __syncthreads();
  float sc = -1e30f;
  if (tid < T_) {
    if (tid < step) {
      const float* row = pre + ((size_t)b * T_ + tid) * H_;
      sc = 0.f;
      for (int k = 0; k < H_; ++k) sc += sh_h[k] * row[k];
    }
    sh_r[tid] = sc;
  }
  __syncthreads();
  for (int off = 32; off > 0; off >>= 1) {
    if (tid < off) sh_r[tid] = fmaxf(sh_r[tid], sh_r[tid + off]);
    __syncthreads();
  }
  float mx = sh_r[0];
  __syncthreads();
  if (tid < T_) {
    float e = (tid < step) ? __expf(sc - mx) : 0.f;
    sh_w[tid] = e;
    sh_r[tid] = e;
  }
  __syncthreads();
  for (int off = 32; off > 0; off >>= 1) {
    if (tid < off) sh_r[tid] += sh_r[tid + off];
    __syncthreads();
  }
  float inv = 1.0f / sh_r[0];
#pragma unroll
  for (int half = 0; half < 2; ++half) {
    int d = tid + half * 256;
    float a = 0.f;
    for (int tp = 0; tp < step; ++tp) a += sh_w[tp] * ref[((size_t)b * T_ + tp) * H_ + d];
    cat2[(size_t)b * 1024 + d] = f2bf(a * inv);   // ctx2 -> left half
  }
}

// ------------- out2 = tanh(cat2 @ Wself^T + b); history update -------------
__global__ __launch_bounds__(128) void self_out_kernel(
    const unsigned short* __restrict__ cat2, const unsigned short* __restrict__ W,
    const float* __restrict__ bias, const float* __restrict__ hvec_f,
    float* __restrict__ pre, float* __restrict__ ref,
    float* __restrict__ href_f, unsigned short* __restrict__ href_bf, int step) {
  __shared__ uint4 shAq[16 * 1024 / 8];
  unsigned short* shA = (unsigned short*)shAq;
  int wave = threadIdx.x >> 5, lane = threadIdx.x & 31;
  int tile = blockIdx.x * 4 + wave;
  int mt = tile >> 5, nt = tile & 31;
  int ll = lane & 15, lh = lane >> 4;
  int tid = threadIdx.x;
#pragma unroll
  for (int cix = tid; cix < 2048; cix += 128) {
    int row = cix >> 7;
    int col = (cix & 127) * 8;
    async_cp16(lds_off_of(&shA[row * 1024 + col]),
               cat2 + (size_t)(mt * 16 + row) * 1024 + col);
  }
  const unsigned short* wrow = W + (size_t)(nt * 16 + ll) * 1024;
  __builtin_prefetch((const void*)wrow, 0, 2);
  async_wait();
  __syncthreads();

  const unsigned short* arow = shA + ll * 1024;
  v8f acc = {};
#pragma unroll
  for (int k0 = 0; k0 < 1024; k0 += 32) {
    v16bf a = load_a(arow, k0, lh);
    v16bf b = load_b(wrow, k0, lh);
    acc = __builtin_amdgcn_wmma_f32_16x16x32_bf16(false, a, false, b, (short)0, acc, false, false);
  }
  int n = nt * 16 + ll;
  float bb = bias[n];
  int mbase = mt * 16 + lh * 8;
#pragma unroll
  for (int r = 0; r < 8; ++r) {
    int m = mbase + r;
    float t = tanhf(acc[r] + bb);                 // step > 0: h_ref = out2
    href_f[(size_t)m * H_ + n] = t;
    href_bf[(size_t)m * H_ + n] = f2bf(t);
    float hv = hvec_f[(size_t)m * H_ + n];
    pre[((size_t)m * T_ + step) * H_ + n] = hv;
    ref[((size_t)m * T_ + step) * H_ + n] = t;
  }
}

// ------------- step 0: h_ref = h_vec (identity), history update ------------
__global__ __launch_bounds__(256) void step0_kernel(
    const float* __restrict__ hvec_f, float* __restrict__ pre, float* __restrict__ ref,
    float* __restrict__ href_f, unsigned short* __restrict__ href_bf) {
  int idx = blockIdx.x * blockDim.x + threadIdx.x;   // B*H
  int b = idx >> 9, d = idx & 511;
  float v = hvec_f[idx];
  href_f[idx] = v;
  href_bf[idx] = f2bf(v);
  pre[((size_t)b * T_ + 0) * H_ + d] = v;
  ref[((size_t)b * T_ + 0) * H_ + d] = v;
}

// --------------------------- prediction head -------------------------------
__global__ __launch_bounds__(256) void head_kernel(
    const float* __restrict__ hin,
    const float* __restrict__ w1, const float* __restrict__ b1,
    const float* __restrict__ w2, const float* __restrict__ b2,
    const float* __restrict__ w3, const float* __restrict__ b3,
    float* __restrict__ out, int t) {
  __shared__ float sin_[H_];
  __shared__ float s1[200];
  __shared__ float s2[50];
  int b = blockIdx.x, tid = threadIdx.x;
  sin_[tid] = hin[b * H_ + tid];
  sin_[tid + 256] = hin[b * H_ + 256 + tid];
  __syncthreads();
  if (tid < 200) {
    float a = b1[tid];
    const float* wr = w1 + tid * H_;
    for (int k = 0; k < H_; ++k) a += wr[k] * sin_[k];
    s1[tid] = fmaxf(a, 0.f);
  }
  __syncthreads();
  if (tid < 50) {
    float a = b2[tid];
    const float* wr = w2 + tid * 200;
    for (int k = 0; k < 200; ++k) a += wr[k] * s1[k];
    s2[tid] = fmaxf(a, 0.f);
  }
  __syncthreads();
  if (tid < O_) {
    float a = b3[tid];
    const float* wr = w3 + tid * 50;
    for (int k = 0; k < 50; ++k) a += wr[k] * s2[k];
    out[((size_t)b * T_ + t) * O_ + tid] = a;
  }
}

// --------------------------- workspace layout (bytes) ----------------------
static const size_t oXBF   = 0;          // B*T*H bf16   4MB
static const size_t oWIH   = 4194304;    // L*G*H bf16   4MB
static const size_t oWHH   = 8388608;    //              4MB
static const size_t oWINT  = 12582912;   // L*H*2H bf16  2MB
static const size_t oWSELF = 14680064;   //              2MB
static const size_t oHRAWB = 16777216;   // L*B*H bf16 128KB
static const size_t oHREFB = 16908288;   // B*H bf16    64KB
static const size_t oCAT1  = 16973824;   // B*2H bf16  128KB
static const size_t oCAT2  = 17104896;   //            128KB
static const size_t oCF    = 17235968;   // L*B*H f32  256KB
static const size_t oGATES = 17498112;   // B*G f32    512KB
static const size_t oHVECF = 18022400;   // B*H f32    128KB
static const size_t oHREFF = 18153472;   // B*H f32    128KB
static const size_t oPRE   = 18284544;   // L*B*T*H f32 16MB
static const size_t oREF   = 35061760;   //             16MB

extern "C" void kernel_launch(void* const* d_in, const int* in_sizes, int n_in,
                              void* d_out, int out_size, void* d_ws, size_t ws_size,
                              hipStream_t stream) {
  const float* x    = (const float*)d_in[0];
  const float* enc  = (const float*)d_in[1];
  const float* h0   = (const float*)d_in[2];
  const float* c0   = (const float*)d_in[3];
  const float* w_ih = (const float*)d_in[4];
  const float* w_hh = (const float*)d_in[5];
  const float* b_ih = (const float*)d_in[6];
  const float* b_hh = (const float*)d_in[7];
  const float* w_it = (const float*)d_in[8];
  const float* b_it = (const float*)d_in[9];
  const float* w_sf = (const float*)d_in[10];
  const float* b_sf = (const float*)d_in[11];
  const float* w1 = (const float*)d_in[12];
  const float* b1 = (const float*)d_in[13];
  const float* w2 = (const float*)d_in[14];
  const float* b2 = (const float*)d_in[15];
  const float* w3 = (const float*)d_in[16];
  const float* b3 = (const float*)d_in[17];
  float* out = (float*)d_out;

  char* ws = (char*)d_ws;
  unsigned short* x_bf     = (unsigned short*)(ws + oXBF);
  unsigned short* wih_bf   = (unsigned short*)(ws + oWIH);
  unsigned short* whh_bf   = (unsigned short*)(ws + oWHH);
  unsigned short* wint_bf  = (unsigned short*)(ws + oWINT);
  unsigned short* wself_bf = (unsigned short*)(ws + oWSELF);
  unsigned short* hraw_bf  = (unsigned short*)(ws + oHRAWB);
  unsigned short* href_bf  = (unsigned short*)(ws + oHREFB);
  unsigned short* cat1 = (unsigned short*)(ws + oCAT1);
  unsigned short* cat2 = (unsigned short*)(ws + oCAT2);
  float* c_f    = (float*)(ws + oCF);
  float* gates  = (float*)(ws + oGATES);
  float* hvec_f = (float*)(ws + oHVECF);
  float* href_f = (float*)(ws + oHREFF);
  float* pre_h  = (float*)(ws + oPRE);
  float* ref_h  = (float*)(ws + oREF);

  // ---- one-time per call: bf16 conversions + state init -------------------
  auto cvt = [&](const float* in, unsigned short* o, int n) {
    cvt_bf16_kernel<<<(n + 255) / 256, 256, 0, stream>>>(in, o, n);
  };
  cvt(x, x_bf, B_ * T_ * H_);
  cvt(w_ih, wih_bf, L_ * G_ * H_);
  cvt(w_hh, whh_bf, L_ * G_ * H_);
  cvt(w_it, wint_bf, L_ * H_ * 2 * H_);
  cvt(w_sf, wself_bf, L_ * H_ * 2 * H_);
  cvt(h0, hraw_bf, L_ * B_ * H_);
  hipMemcpyAsync(c_f, c0, (size_t)L_ * B_ * H_ * sizeof(float),
                 hipMemcpyDeviceToDevice, stream);

  // ---- sequential decode loop --------------------------------------------
  for (int t = 0; t < T_; ++t) {
    for (int l = 0; l < L_; ++l) {
      const unsigned short* in_bf = (l == 0) ? (x_bf + (size_t)t * H_) : href_bf;
      int lda = (l == 0) ? (T_ * H_) : H_;
      unsigned short* wih_l = wih_bf + (size_t)l * G_ * H_;
      unsigned short* whh_l = whh_bf + (size_t)l * G_ * H_;
      unsigned short* wit_l = wint_bf + (size_t)l * H_ * 1024;
      unsigned short* wsf_l = wself_bf + (size_t)l * H_ * 1024;
      unsigned short* hrb_l = hraw_bf + (size_t)l * B_ * H_;
      float* c_l   = c_f + (size_t)l * B_ * H_;
      float* pre_l = pre_h + (size_t)l * B_ * T_ * H_;
      float* ref_l = ref_h + (size_t)l * B_ * T_ * H_;

      gates_kernel<<<128, 128, 0, stream>>>(in_bf, lda, hrb_l, wih_l, whh_l,
                                            b_ih + l * G_, b_hh + l * G_, gates);
      cell_attn_kernel<<<64, 256, 0, stream>>>(gates, c_l, hrb_l, cat1, enc);
      hvec_kernel<<<32, 128, 0, stream>>>(cat1, wit_l, b_it + l * H_, hvec_f, cat2);
      if (t == 0) {
        step0_kernel<<<128, 256, 0, stream>>>(hvec_f, pre_l, ref_l, href_f, href_bf);
      } else {
        self_attn_kernel<<<64, 256, 0, stream>>>(hvec_f, pre_l, ref_l, cat2, t);
        self_out_kernel<<<32, 128, 0, stream>>>(cat2, wsf_l, b_sf + l * H_, hvec_f,
                                                pre_l, ref_l, href_f, href_bf, t);
      }
    }
    head_kernel<<<64, 256, 0, stream>>>(href_f, w1, b1, w2, b2, w3, b3, out, t);
  }
}
